// OnlineDFlashModel_57406532878477
// MI455X (gfx1250) — compile-verified, matrix operand order checked by reading, not run on hardware
//
#include <hip/hip_runtime.h>
#include <hip/hip_bf16.h>

typedef __attribute__((ext_vector_type(16))) _Float16 v16h;
typedef __attribute__((ext_vector_type(8)))  _Float16 v8h;
typedef __attribute__((ext_vector_type(8)))  float    v8f;

#define L_SEQ 1024
#define D_MODEL 2048
#define N_HEADS 16
#define HD 128
#define VOCAB 32000
#define MASK_ID 31999

// ---------------------------------------------------------------------------
// Weight convert + transpose: out[n*K + k] = (f16) in[k*N + n]
// ---------------------------------------------------------------------------
__global__ void __launch_bounds__(256)
cvt_transpose_f32_f16(const float* __restrict__ in, _Float16* __restrict__ out,
                      int K, int N) {
  long long idx = (long long)blockIdx.x * blockDim.x + threadIdx.x;
  long long total = (long long)K * N;
  if (idx >= total) return;
  int n = (int)(idx % N);
  long long k = idx / N;
  out[(long long)n * K + k] = (_Float16)in[k * (long long)N + n];
}

// ---------------------------------------------------------------------------
// Build noise embeddings (gather) + f16 kv_in = [hidden_states ; noise_emb]
// ---------------------------------------------------------------------------
__global__ void __launch_bounds__(256)
build_inputs_kernel(const int* __restrict__ ids, const float* __restrict__ hs,
                    const float* __restrict__ emb,
                    _Float16* __restrict__ noise, _Float16* __restrict__ kv) {
  long long idx = (long long)blockIdx.x * blockDim.x + threadIdx.x;
  if (idx >= (long long)L_SEQ * D_MODEL) return;
  int d   = (int)(idx % D_MODEL);
  int pos = (int)(idx / D_MODEL);
  int tok = ((pos & 15) == 0) ? ids[pos] : MASK_ID;
  _Float16 ne = (_Float16)emb[(long long)tok * D_MODEL + d];
  noise[idx] = ne;
  kv[idx] = (_Float16)hs[idx];
  kv[(long long)L_SEQ * D_MODEL + idx] = ne;
}

// ---------------------------------------------------------------------------
// WMMA GEMM, block tile 128(M) x 64(N) x 32(K):
//   - 8 waves / block; wave w owns rows [tm*16, tm*16+16), tm = tmB*8 + w
//   - each wave computes 16x64 via 4 accumulators (A fragment reused 4x)
//   - B panel (64 cols x 32 K, f16, 4KB) staged in LDS via
//     global_load_async_to_lds_b128 (ASYNCcnt), double-buffered
//   - all 4 B fragments preloaded (ds clause), then 4 back-to-back WMMAs
// A is MxK row-major f16; Bt is NxK row-major f16 (i.e. B pre-transposed).
// CMODE: 0 = f32 store, 1 = f16 store, 2 = f16 transposed store (C[n*ldc+m])
// Batched over gridDim.y with element strides. All dims must divide evenly.
// ---------------------------------------------------------------------------
#define ASYNC_B128(ldsoff, gaddr)                                          \
  asm volatile("global_load_async_to_lds_b128 %0, %1, off"                 \
               :: "v"(ldsoff), "v"(gaddr) : "memory")
#define WAIT_ASYNC() asm volatile("s_wait_asynccnt 0x0" ::: "memory")

template<int CMODE>
__global__ void __launch_bounds__(256)
gemm_wmma_f16(const _Float16* __restrict__ A, long long strideA, int lda,
              const _Float16* __restrict__ Bt, long long strideB, int ldb,
              void* __restrict__ Cv, long long strideC, int ldc,
              int M, int N, int K) {
  __shared__ __align__(16) _Float16 panel[2][64 * 32];

  const int tid  = threadIdx.x;
  const int wave = tid >> 5;
  const int lane = tid & 31;
  const int nBlocks = N >> 6;                       // N / 64
  const int tmB = blockIdx.x / nBlocks;
  const int tnB = blockIdx.x - tmB * nBlocks;
  const long long batch = blockIdx.y;

  const _Float16* __restrict__ Ab = A  + batch * strideA;
  const _Float16* __restrict__ Bb = Bt + batch * strideB;

  // B panel staging: thread t -> (n_off = t/4, k_off = (t%4)*8), one b128 each
  const int n_off = tid >> 2;
  const int k_off = (tid & 3) << 3;
  const _Float16* bsrc = Bb + (long long)(tnB * 64 + n_off) * ldb + k_off;
  const unsigned ldsdst0 =
      (unsigned)(size_t)&panel[0][n_off * 32 + k_off];  // low 32b = LDS offset
  const unsigned ldsdst1 =
      (unsigned)(size_t)&panel[1][n_off * 32 + k_off];

  // A fragment addressing (ISA 16-bit operand lane layout)
  const int r   = lane & 15;
  const int sel = (lane >> 4) << 3;                 // 0 or 8
  const _Float16* arow =
      Ab + (long long)((tmB * 8 + wave) * 16 + r) * lda;

  v8f acc0 = {}, acc1 = {}, acc2 = {}, acc3 = {};

  // preload panel 0
  ASYNC_B128(ldsdst0, (unsigned long long)(const void*)bsrc);
  WAIT_ASYNC();
  __syncthreads();

  const int nk = K >> 5;
  for (int ks = 0; ks < nk; ++ks) {
    const int k0 = ks << 5;
    // branch-free prefetch of next panel: last iteration re-fetches (ignored,
    // writes the panel not being read; prefetch is speculative-safe)
    {
      const int knext = (ks + 1 < nk) ? (ks + 1) : (nk - 1);
      const unsigned long long g =
          (unsigned long long)(const void*)(bsrc + (long long)knext * 32);
      ASYNC_B128(((ks + 1) & 1) ? ldsdst1 : ldsdst0, g);
      __builtin_prefetch(arow + (long long)knext * 32, 0, 0);
    }
    const _Float16* pb = panel[ks & 1];

    // A fragment (2 x global b128)
    v8h a0 = *(const v8h*)(arow + k0 + sel);
    v8h a1 = *(const v8h*)(arow + k0 + 16 + sel);
    v16h a;
#pragma unroll
    for (int i = 0; i < 8; ++i) { a[i] = a0[i]; a[i + 8] = a1[i]; }

    // All four B fragments first (8 x ds_load_b128, distinct registers)
    v16h bf0, bf1, bf2, bf3;
#define LOAD_B(BF, C)                                                       \
    {                                                                       \
      const _Float16* col = pb + ((C) * 16 + r) * 32;                       \
      v8h b0 = *(const v8h*)(col + sel);                                    \
      v8h b1 = *(const v8h*)(col + 16 + sel);                               \
      _Pragma("unroll")                                                     \
      for (int i = 0; i < 8; ++i) { BF[i] = b0[i]; BF[i + 8] = b1[i]; }     \
    }
    LOAD_B(bf0, 0)
    LOAD_B(bf1, 1)
    LOAD_B(bf2, 2)
    LOAD_B(bf3, 3)
#undef LOAD_B

    // 4 back-to-back WMMAs (independent accumulators: no D->A/B hazard)
    acc0 = __builtin_amdgcn_wmma_f32_16x16x32_f16(false, a, false, bf0,
                                                  (short)0, acc0, false, false);
    acc1 = __builtin_amdgcn_wmma_f32_16x16x32_f16(false, a, false, bf1,
                                                  (short)0, acc1, false, false);
    acc2 = __builtin_amdgcn_wmma_f32_16x16x32_f16(false, a, false, bf2,
                                                  (short)0, acc2, false, false);
    acc3 = __builtin_amdgcn_wmma_f32_16x16x32_f16(false, a, false, bf3,
                                                  (short)0, acc3, false, false);

    WAIT_ASYNC();
    __syncthreads();
  }

  // store: C layout -> M = cm0 + i (i = vgpr), N = cnb + c*16
  const int cm0 = (tmB * 8 + wave) * 16 + sel;
  const int cnb = tnB * 64 + (lane & 15);
  if (CMODE == 0) {
    float* C = (float*)Cv + batch * strideC;
#define ST(ACC, C_)                                                         \
    _Pragma("unroll")                                                       \
    for (int i = 0; i < 8; ++i)                                             \
      C[(long long)(cm0 + i) * ldc + cnb + (C_) * 16] = ACC[i];
    ST(acc0, 0) ST(acc1, 1) ST(acc2, 2) ST(acc3, 3)
#undef ST
  } else if (CMODE == 1) {
    _Float16* C = (_Float16*)Cv + batch * strideC;
#define ST(ACC, C_)                                                         \
    _Pragma("unroll")                                                       \
    for (int i = 0; i < 8; ++i)                                             \
      C[(long long)(cm0 + i) * ldc + cnb + (C_) * 16] = (_Float16)ACC[i];
    ST(acc0, 0) ST(acc1, 1) ST(acc2, 2) ST(acc3, 3)
#undef ST
  } else {
    _Float16* C = (_Float16*)Cv + batch * strideC;
#define ST(ACC, C_)                                                         \
    _Pragma("unroll")                                                       \
    for (int i = 0; i < 8; ++i)                                             \
      C[(long long)(cnb + (C_) * 16) * ldc + (cm0 + i)] = (_Float16)ACC[i];
    ST(acc0, 0) ST(acc1, 1) ST(acc2, 2) ST(acc3, 3)
#undef ST
  }
}

// ---------------------------------------------------------------------------
// Masked softmax over one (head, q) score row of 2L keys -> f16 probs
// ---------------------------------------------------------------------------
__global__ void __launch_bounds__(256)
softmax_mask_kernel(const float* __restrict__ S, _Float16* __restrict__ P) {
  const int q = blockIdx.x;
  const int h = blockIdx.y;
  const int tid = threadIdx.x;
  const int KV = 2 * L_SEQ;
  const float* row = S + ((long long)h * L_SEQ + q) * (long long)KV;
  _Float16*  prow = P + ((long long)h * L_SEQ + q) * (long long)KV;
  const int qb = q >> 4;
  const float scale = 0.08838834764831845f;     // 1/sqrt(128)
  __shared__ float red[256];

  float m = -3.0e38f;
  for (int j = tid; j < KV; j += 256) {
    bool vis = (j < L_SEQ) ? ((j >> 4) < qb) : (((j - L_SEQ) >> 4) == qb);
    if (vis) { float x = row[j] * scale; m = fmaxf(m, x); }
  }
  red[tid] = m; __syncthreads();
  for (int s = 128; s > 0; s >>= 1) {
    if (tid < s) red[tid] = fmaxf(red[tid], red[tid + s]);
    __syncthreads();
  }
  m = red[0]; __syncthreads();

  float sum = 0.f;
  for (int j = tid; j < KV; j += 256) {
    bool vis = (j < L_SEQ) ? ((j >> 4) < qb) : (((j - L_SEQ) >> 4) == qb);
    float e = vis ? __expf(row[j] * scale - m) : 0.f;
    prow[j] = (_Float16)e;
    sum += e;
  }
  red[tid] = sum; __syncthreads();
  for (int s = 128; s > 0; s >>= 1) {
    if (tid < s) red[tid] += red[tid + s];
    __syncthreads();
  }
  const float inv = 1.0f / red[0];
  for (int j = tid; j < KV; j += 256) {
    prow[j] = (_Float16)((float)prow[j] * inv);
  }
}

// ---------------------------------------------------------------------------
// Per-row CE + argmax; atomics into acc[0]=ce_sum, acc[1]=correct, acc[2]=count
// ---------------------------------------------------------------------------
__global__ void __launch_bounds__(256)
ce_argmax_kernel(const float* __restrict__ logits, const int* __restrict__ ids,
                 const float* __restrict__ loss_mask, float* __restrict__ acc) {
  const int pos = blockIdx.x;
  const int tid = threadIdx.x;
  const float* row = logits + (long long)pos * VOCAB;
  __shared__ float rmax[256];
  __shared__ int   ridx[256];

  float m = -3.0e38f; int mi = 0;
  for (int j = tid; j < VOCAB; j += 256) {
    float x = row[j];
    if (x > m) { m = x; mi = j; }
  }
  rmax[tid] = m; ridx[tid] = mi; __syncthreads();
  for (int s = 128; s > 0; s >>= 1) {
    if (tid < s) {
      float om = rmax[tid + s]; int oi = ridx[tid + s];
      if (om > rmax[tid] || (om == rmax[tid] && oi < ridx[tid])) {
        rmax[tid] = om; ridx[tid] = oi;
      }
    }
    __syncthreads();
  }
  m = rmax[0];
  const int pred = ridx[0];
  __syncthreads();

  float sum = 0.f;
  for (int j = tid; j < VOCAB; j += 256) sum += __expf(row[j] - m);
  rmax[tid] = sum; __syncthreads();
  for (int s = 128; s > 0; s >>= 1) {
    if (tid < s) rmax[tid] += rmax[tid + s];
    __syncthreads();
  }
  if (tid == 0) {
    int t = ids[pos];
    float ce = -(row[t] - m - __logf(rmax[0]));
    bool valid = (pos >= 16) && ((pos & 15) != 0);
    float combined = loss_mask[pos] * (valid ? 1.f : 0.f);
    if (combined > 1e-6f) {
      atomicAdd(&acc[0], ce);
      atomicAdd(&acc[1], (pred == t) ? 1.f : 0.f);
      atomicAdd(&acc[2], 1.f);
    }
  }
}

__global__ void init_acc_kernel(float* acc) {
  if (threadIdx.x < 3) acc[threadIdx.x] = 0.f;
}

__global__ void finalize_kernel(const float* __restrict__ acc, float* __restrict__ out) {
  if (threadIdx.x == 0) {
    float cnt = acc[2];
    out[0] = acc[0] / cnt;   // loss
    out[1] = acc[1] / cnt;   // accuracy
  }
}

// ---------------------------------------------------------------------------
// Host side
// ---------------------------------------------------------------------------
static inline long long alignUp(long long x) { return (x + 255) & ~255LL; }

extern "C" void kernel_launch(void* const* d_in, const int* in_sizes, int n_in,
                              void* d_out, int out_size, void* d_ws, size_t ws_size,
                              hipStream_t stream) {
  (void)in_sizes; (void)n_in; (void)out_size; (void)ws_size;
  const int*   input_ids  = (const int*)  d_in[0];
  // d_in[1] = attention_mask (unused: all ones)
  const float* hidden_st  = (const float*)d_in[2];
  const float* loss_mask  = (const float*)d_in[3];
  const float* embed_tab  = (const float*)d_in[4];
  const float* Wq         = (const float*)d_in[5];
  const float* Wk         = (const float*)d_in[6];
  const float* Wv         = (const float*)d_in[7];
  const float* Wo         = (const float*)d_in[8];
  const float* lm_head    = (const float*)d_in[9];

  char* ws = (char*)d_ws;
  long long off = 0;
  auto take = [&](long long bytes) { long long o = off; off = alignUp(off + bytes); return o; };

  const long long DD   = (long long)D_MODEL * D_MODEL;
  const long long LD   = (long long)L_SEQ * D_MODEL;
  const long long KVD  = 2LL * LD;
  const long long HLK  = (long long)N_HEADS * L_SEQ * 2 * L_SEQ;

  _Float16* wq_t  = (_Float16*)(ws + take(DD * 2));
  _Float16* wk_t  = (_Float16*)(ws + take(DD * 2));
  _Float16* wv_t  = (_Float16*)(ws + take(DD * 2));
  _Float16* wo_t  = (_Float16*)(ws + take(DD * 2));
  _Float16* lm_t  = (_Float16*)(ws + take((long long)VOCAB * D_MODEL * 2));
  _Float16* noise = (_Float16*)(ws + take(LD * 2));
  _Float16* kv    = (_Float16*)(ws + take(KVD * 2));
  _Float16* qb    = (_Float16*)(ws + take(LD * 2));
  _Float16* kb    = (_Float16*)(ws + take(KVD * 2));
  _Float16* vt    = (_Float16*)(ws + take(KVD * 2));
  _Float16* attn  = (_Float16*)(ws + take(HLK * 2));
  _Float16* hin   = (_Float16*)(ws + take(LD * 2));
  _Float16* hid   = (_Float16*)(ws + take(LD * 2));
  float*    scores= (float*)   (ws + take(HLK * 4));   // reused for logits
  float*    accum = (float*)   (ws + take(16));
  float*    logits = scores;   // scores dead after softmax; 131MB <= 134MB

  const int T = 256;
  auto gblk = [&](long long n) { return (unsigned)((n + T - 1) / T); };

  // 1) weight convert + transpose
  cvt_transpose_f32_f16<<<gblk(DD), T, 0, stream>>>(Wq, wq_t, D_MODEL, D_MODEL);
  cvt_transpose_f32_f16<<<gblk(DD), T, 0, stream>>>(Wk, wk_t, D_MODEL, D_MODEL);
  cvt_transpose_f32_f16<<<gblk(DD), T, 0, stream>>>(Wv, wv_t, D_MODEL, D_MODEL);
  cvt_transpose_f32_f16<<<gblk(DD), T, 0, stream>>>(Wo, wo_t, D_MODEL, D_MODEL);
  cvt_transpose_f32_f16<<<gblk((long long)VOCAB * D_MODEL), T, 0, stream>>>(
      lm_head, lm_t, D_MODEL, VOCAB);

  // 2) gather + f16 inputs
  build_inputs_kernel<<<gblk(LD), T, 0, stream>>>(input_ids, hidden_st, embed_tab, noise, kv);

  // block tile = 128(M) x 64(N)
  auto gemmBlocks = [&](int M, int N) { return (unsigned)((M / 128) * (N / 64)); };

  // 3) q = noise @ Wq   (1024x2048x2048) -> f16
  gemm_wmma_f16<1><<<gemmBlocks(L_SEQ, D_MODEL), T, 0, stream>>>(
      noise, 0, D_MODEL, wq_t, 0, D_MODEL, qb, 0, D_MODEL, L_SEQ, D_MODEL, D_MODEL);
  // 4) k = kv @ Wk      (2048x2048x2048) -> f16
  gemm_wmma_f16<1><<<gemmBlocks(2 * L_SEQ, D_MODEL), T, 0, stream>>>(
      kv, 0, D_MODEL, wk_t, 0, D_MODEL, kb, 0, D_MODEL, 2 * L_SEQ, D_MODEL, D_MODEL);
  // 5) v = kv @ Wv -> stored transposed: vt[(h*128+d)*2048 + key]
  gemm_wmma_f16<2><<<gemmBlocks(2 * L_SEQ, D_MODEL), T, 0, stream>>>(
      kv, 0, D_MODEL, wv_t, 0, D_MODEL, vt, 0, 2 * L_SEQ, 2 * L_SEQ, D_MODEL, D_MODEL);

  // 6) scores[h] = q_h @ k_h^T  (1024 x 2048 x 128), batched over heads -> f32
  {
    dim3 grid(gemmBlocks(L_SEQ, 2 * L_SEQ), N_HEADS);
    gemm_wmma_f16<0><<<grid, T, 0, stream>>>(
        qb, HD, D_MODEL, kb, HD, D_MODEL,
        scores, (long long)L_SEQ * 2 * L_SEQ, 2 * L_SEQ,
        L_SEQ, 2 * L_SEQ, HD);
  }

  // 7) masked softmax -> f16 probs
  {
    dim3 grid(L_SEQ, N_HEADS);
    softmax_mask_kernel<<<grid, T, 0, stream>>>(scores, attn);
  }

  // 8) ctx[h] = attn_h @ v_h (1024 x 128 x 2048), batched; into hin columns
  {
    dim3 grid(gemmBlocks(L_SEQ, HD), N_HEADS);
    gemm_wmma_f16<1><<<grid, T, 0, stream>>>(
        attn, (long long)L_SEQ * 2 * L_SEQ, 2 * L_SEQ,
        vt, (long long)HD * 2 * L_SEQ, 2 * L_SEQ,
        hin, HD, D_MODEL,
        L_SEQ, HD, 2 * L_SEQ);
  }

  // 9) hidden = ctx @ Wo (1024x2048x2048) -> f16
  gemm_wmma_f16<1><<<gemmBlocks(L_SEQ, D_MODEL), T, 0, stream>>>(
      hin, 0, D_MODEL, wo_t, 0, D_MODEL, hid, 0, D_MODEL, L_SEQ, D_MODEL, D_MODEL);

  // 10) logits = hidden @ lm_head (1024x32000x2048) -> f32 (reuse scores buf)
  gemm_wmma_f16<0><<<gemmBlocks(L_SEQ, VOCAB), T, 0, stream>>>(
      hid, 0, D_MODEL, lm_t, 0, D_MODEL, logits, 0, VOCAB, L_SEQ, VOCAB, D_MODEL);

  // 11) CE + accuracy
  init_acc_kernel<<<1, 32, 0, stream>>>(accum);
  ce_argmax_kernel<<<L_SEQ, T, 0, stream>>>(logits, input_ids, loss_mask, accum);
  finalize_kernel<<<1, 32, 0, stream>>>(accum, (float*)d_out);
}
